// GPT_56281251447184
// MI455X (gfx1250) — matrix operand order repaired and checked by reading, not compiled, not run on hardware
//
#include <hip/hip_runtime.h>
#include <hip/hip_bf16.h>

typedef __attribute__((ext_vector_type(16))) _Float16 v16h;
typedef __attribute__((ext_vector_type(8)))  _Float16 v8h;
typedef __attribute__((ext_vector_type(8)))  float    v8f;

#define L_  8
#define H_  16
#define E_  1024
#define V_  32000
#define B_  4
#define T_  1024
#define HS_ 64
#define M_  (B_*T_)   /* 4096 tokens */

// ---------------------------------------------------------------------------
// CDNA5 async global->LDS copy (ASYNCcnt-tracked, no VGPR round-trip).
// LDS address comes from the vdst VGPR; GV mode: 64-bit address in VGPR pair.
// ---------------------------------------------------------------------------
__device__ inline void async_copy_b128(unsigned int lds_off, const void* gptr) {
  asm volatile("global_load_async_to_lds_b128 %0, %1, off"
               :: "v"(lds_off), "v"(gptr) : "memory");
}
__device__ inline void wait_async_le2() {
  asm volatile("s_wait_asynccnt 0x2" ::: "memory");
}
__device__ inline void wait_async_0() {
  asm volatile("s_wait_asynccnt 0x0" ::: "memory");
}

// ---------------------------------------------------------------------------
// Fragment loaders (this code's contiguous k order for WMMA 16x16x32 f16, wave32)
//  A 16x32 (MxK): lane(<16)=row M, holds K=0..15 (h0..15);
//                 lane(>=16)=row M, holds K=16..31.
//  B 32x16 (KxN): lane = N + 16*(K>=16); half h -> K = h + 16*(lane>=16).
//  C/D 16x16:     VGPR r: lanes0-15 -> (M=r, N=lane); lanes16-31 -> (M=8+r, N=lane-16)
// ---------------------------------------------------------------------------
__device__ inline v16h load_a_frag(const _Float16* p, int stride) {
  const int lane = threadIdx.x & 31;
  const int row  = lane & 15;
  const int klo  = (lane >> 4) << 4;            // 0 or 16
  const _Float16* rp = p + (size_t)row * stride + klo;
  v8h lo = *(const v8h*)(rp);                    // K = klo .. klo+7
  v8h hi = *(const v8h*)(rp + 8);                // K = klo+8 .. klo+15
  v16h r;
  #pragma unroll
  for (int i = 0; i < 8; i++) { r[i] = lo[i]; r[i + 8] = hi[i]; }
  return r;
}

__device__ inline v16h load_b_fragT(const _Float16* p, int stride) {
  // B[k][n] where memory is row-major [n][k] (e.g. K-matrix rows = keys)
  const int lane = threadIdx.x & 31;
  const int n    = lane & 15;
  const int khi  = (lane >> 4) << 4;             // 0 or 16
  const _Float16* rp = p + (size_t)n * stride + khi;
  v8h lo = *(const v8h*)(rp);
  v8h hi = *(const v8h*)(rp + 8);
  v16h r;
  #pragma unroll
  for (int i = 0; i < 8; i++) { r[i] = lo[i]; r[i + 8] = hi[i]; }
  return r;
}

__device__ inline v16h load_b_frag(const _Float16* p, int stride) {
  // B[k][n], memory row-major [k][n] (e.g. V-matrix) -> strided gather
  const int lane = threadIdx.x & 31;
  const int n    = lane & 15;
  const int khi  = (lane >> 4) << 4;
  v16h r;
  #pragma unroll
  for (int h = 0; h < 16; h++) r[h] = p[(size_t)(khi + h) * stride + n];
  return r;
}

__device__ inline float half_max16(float v) {   // reduce over 16 lanes of a half-wave
  #pragma unroll
  for (int off = 1; off < 16; off <<= 1) v = fmaxf(v, __shfl_xor(v, off, 32));
  return v;
}
__device__ inline float half_sum16(float v) {
  #pragma unroll
  for (int off = 1; off < 16; off <<= 1) v += __shfl_xor(v, off, 32);
  return v;
}

// ---------------------------------------------------------------------------
// Embedding: x[b,t,:] = tok_emb[idx[b,t],:] + pos_emb[t,:]
// ---------------------------------------------------------------------------
__global__ __launch_bounds__(256) void embed_k(
    const int* __restrict__ idx, const float* __restrict__ tok,
    const float* __restrict__ pos, float* __restrict__ x) {
  const int row = blockIdx.x;                    // 0..4095  (b*T + t)
  const int t   = row & (T_ - 1);
  const int id  = idx[row];
  const float* te = tok + (size_t)id * E_;
  const float* pe = pos + (size_t)t  * E_;
  float* xr = x + (size_t)row * E_;
  for (int e = threadIdx.x; e < E_; e += 256) xr[e] = te[e] + pe[e];
}

// ---------------------------------------------------------------------------
// LayerNorm (fp32 in) -> f16 out for WMMA consumption
// ---------------------------------------------------------------------------
__global__ __launch_bounds__(256) void layernorm_f16_k(
    const float* __restrict__ x, const float* __restrict__ g,
    const float* __restrict__ b, _Float16* __restrict__ out) {
  __shared__ float red[256];
  const int row = blockIdx.x;
  const float* xr = x + (size_t)row * E_;
  float v[4], s = 0.f, s2 = 0.f;
  #pragma unroll
  for (int i = 0; i < 4; i++) {
    v[i] = xr[threadIdx.x + i * 256];
    s += v[i]; s2 += v[i] * v[i];
  }
  red[threadIdx.x] = s; __syncthreads();
  for (int off = 128; off > 0; off >>= 1) {
    if (threadIdx.x < off) red[threadIdx.x] += red[threadIdx.x + off];
    __syncthreads();
  }
  const float mean = red[0] * (1.0f / E_);
  __syncthreads();
  red[threadIdx.x] = s2; __syncthreads();
  for (int off = 128; off > 0; off >>= 1) {
    if (threadIdx.x < off) red[threadIdx.x] += red[threadIdx.x + off];
    __syncthreads();
  }
  const float var  = red[0] * (1.0f / E_) - mean * mean;
  const float rstd = rsqrtf(var + 1e-5f);
  #pragma unroll
  for (int i = 0; i < 4; i++) {
    const int c = threadIdx.x + i * 256;
    out[(size_t)row * E_ + c] = (_Float16)((v[i] - mean) * rstd * g[c] + b[c]);
  }
}

// ---------------------------------------------------------------------------
// Templated WMMA GEMM: out = [relu]( A(f16)[M,K] * W(f32->f16)[K,N] + bias + resid )
// 256 thr = 8 waves, tile 128x128, wave = 32x64 (2x4 wmma tiles).
//  - A tile: async DMA (global_load_async_to_lds_b128) into double-buffered
//    row-major LDS; fragments read directly from row-major (stride 32).
//  - B tile: software-pipelined register prefetch + f32->f16 VALU convert into
//    fragment-ready LDS layout.
// Compile-time flags -> straight-line epilogue (no per-element branches).
// ---------------------------------------------------------------------------
template<bool HAS_BIAS, bool HAS_RES, bool DO_RELU, bool OUT_F, bool OUT_H>
__global__ __launch_bounds__(256) void gemm_t(
    const _Float16* __restrict__ A, const float* __restrict__ W,
    const float* __restrict__ bias, const float* __restrict__ resid,
    float* __restrict__ outF, _Float16* __restrict__ outH,
    int M, int N, int K) {
  __shared__ _Float16 sA[2][128 * 32];  // double-buffered, row-major [row][k]
  __shared__ _Float16 sB[4096];         // 8 n-subtiles * 32 lanes * 16 halves
  const int t    = threadIdx.x;
  const int lane = t & 31;
  const int wave = t >> 5;
  const int wm   = wave & 3;            // m: 4 waves * 32 rows
  const int wn   = wave >> 2;           // n: 2 waves * 64 cols
  const int m0   = blockIdx.y * 128;
  const int n0   = blockIdx.x * 128;

  // A async staging: 512 16B chunks, 2 per thread (contiguous 32B)
  const int achunk = t * 2;             // chunk id of first chunk
  const int arow   = achunk >> 2;       // 0..127
  const int acoff  = (achunk & 3) * 8;  // halves within row: 0 or 16
  const unsigned int aldsoff[2] = {
      (unsigned int)(size_t)&sA[0][arow * 32 + acoff],
      (unsigned int)(size_t)&sA[1][arow * 32 + acoff]};
  const _Float16* agp = A + (size_t)(m0 + arow) * K + acoff;

  // B staging: thread reads 16 consecutive floats of one W row
  const int brow = t >> 3;              // 0..31 (k within tile)
  const int bn0  = (t & 7) * 16;        // 0..112
  const int bsub = t & 7;
  const int blb  = (brow >= 16) ? 16 : 0;
  const int bhh  = brow & 15;
  const float* wrow = W + (size_t)brow * N + n0 + bn0;

  v8f acc[2][4];
  #pragma unroll
  for (int i = 0; i < 2; i++)
    #pragma unroll
    for (int j = 0; j < 4; j++)
      #pragma unroll
      for (int r = 0; r < 8; r++) acc[i][j][r] = 0.f;

  // ---- prologue: kick off first A tile DMA + first B register load
  async_copy_b128(aldsoff[0], (const void*)agp);
  async_copy_b128(aldsoff[0] + 16, (const void*)(agp + 8));
  float rb[16];
  {
    const float4* wp4 = (const float4*)wrow;
    #pragma unroll
    for (int q = 0; q < 4; q++) {
      float4 f = wp4[q];
      rb[q*4+0] = f.x; rb[q*4+1] = f.y; rb[q*4+2] = f.z; rb[q*4+3] = f.w;
    }
  }

  const int nIter = K >> 5;
  for (int it = 0; it < nIter; ++it) {
    const int buf = it & 1;
    __syncthreads();                       // prev tiles fully consumed
    // store B regs (tile `it`) into fragment-ready LDS (f32->f16)
    #pragma unroll
    for (int i = 0; i < 16; i++)
      sB[bsub * 512 + (i + blb) * 16 + bhh] = (_Float16)rb[i];
    if (it + 1 < nIter) {
      // prefetch next B tile into regs (global loads issue early)
      const float4* wp4 = (const float4*)(wrow + (size_t)(it + 1) * 32 * N);
      #pragma unroll
      for (int q = 0; q < 4; q++) {
        float4 f = wp4[q];
        rb[q*4+0] = f.x; rb[q*4+1] = f.y; rb[q*4+2] = f.z; rb[q*4+3] = f.w;
      }
      // kick off next A tile DMA into the other buffer
      const _Float16* ag = agp + (it + 1) * 32;
      async_copy_b128(aldsoff[buf ^ 1], (const void*)ag);
      async_copy_b128(aldsoff[buf ^ 1] + 16, (const void*)(ag + 8));
      wait_async_le2();                    // tile `it` (older 2 chunks) done
    } else {
      wait_async_0();
    }
    __syncthreads();                       // everyone's A DMA + B stores visible

    v16h af[2], bf[4];
    #pragma unroll
    for (int i = 0; i < 2; i++)
      af[i] = load_a_frag(&sA[buf][(wm * 2 + i) * 16 * 32], 32);
    #pragma unroll
    for (int j = 0; j < 4; j++)
      bf[j] = *(const v16h*)&sB[(wn * 4 + j) * 512 + lane * 16];
    #pragma unroll
    for (int i = 0; i < 2; i++)
      #pragma unroll
      for (int j = 0; j < 4; j++)
        acc[i][j] = __builtin_amdgcn_wmma_f32_16x16x32_f16(
            false, af[i], false, bf[j], (short)0, acc[i][j], false, false);
  }

  // ---- straight-line epilogue (flags are compile-time)
  const int laneHi = lane >> 4;
  const int ncol   = lane & 15;
  #pragma unroll
  for (int i = 0; i < 2; i++)
    #pragma unroll
    for (int j = 0; j < 4; j++) {
      const int gn = n0 + wn * 64 + j * 16 + ncol;
      float bv = 0.f;
      if (HAS_BIAS) bv = bias[gn];
      #pragma unroll
      for (int r = 0; r < 8; r++) {
        const int gm = m0 + wm * 32 + i * 16 + r + 8 * laneHi;
        const size_t o = (size_t)gm * N + gn;
        float v = acc[i][j][r] + bv;
        if (HAS_RES) v += resid[o];
        if (DO_RELU) v = fmaxf(v, 0.f);
        if (OUT_F)   outF[o] = v;
        if (OUT_H)   outH[o] = (_Float16)v;
      }
    }
}

// ---------------------------------------------------------------------------
// Flash attention (causal), scale = E^-0.5 = 1/32 (matches reference!).
// One wave handles one (b, h, 16-row q tile); 32-key inner tiles.
// ---------------------------------------------------------------------------
__global__ __launch_bounds__(256) void attn_flash_k(
    const _Float16* __restrict__ q16, const _Float16* __restrict__ k16,
    const _Float16* __restrict__ v16, _Float16* __restrict__ o16) {
  __shared__ _Float16 pbuf[8 * 512];             // per-wave 16x32 P staging
  const int lane = threadIdx.x & 31;
  const int wave = threadIdx.x >> 5;
  const int wid  = blockIdx.x * 8 + wave;        // 0..4095
  const int qt   = wid & 63;
  const int bh   = wid >> 6;
  const int bb   = bh >> 4;
  const int hh   = bh & 15;
  const int q0   = qt * 16;
  const int headOff = hh * HS_;
  const size_t tokBase = (size_t)bb * T_;
  const int laneHi = lane >> 4;
  const int ncol   = lane & 15;
  _Float16* pw = &pbuf[wave * 512];

  v16h aq[2];
  #pragma unroll
  for (int ks = 0; ks < 2; ks++)
    aq[ks] = load_a_frag(q16 + (tokBase + q0) * E_ + headOff + ks * 32, E_);

  v8f oacc[4];
  #pragma unroll
  for (int n = 0; n < 4; n++)
    #pragma unroll
    for (int r = 0; r < 8; r++) oacc[n][r] = 0.f;
  float mrow[8], lrow[8];
  #pragma unroll
  for (int r = 0; r < 8; r++) { mrow[r] = -__builtin_inff(); lrow[r] = 0.f; }

  const float sc = 0.03125f;                     // 1/sqrt(E)
  for (int j0 = 0; j0 <= q0; j0 += 32) {         // every row keeps >=1 valid key
    v8f s[2];
    #pragma unroll
    for (int hlf = 0; hlf < 2; hlf++) {
      #pragma unroll
      for (int r = 0; r < 8; r++) s[hlf][r] = 0.f;
      const _Float16* kb = k16 + (tokBase + j0 + hlf * 16) * E_ + headOff;
      v16h b0 = load_b_fragT(kb, E_);            // K^T, d = 0..31
      v16h b1 = load_b_fragT(kb + 32, E_);       // d = 32..63
      s[hlf] = __builtin_amdgcn_wmma_f32_16x16x32_f16(false, aq[0], false, b0,
                                                      (short)0, s[hlf], false, false);
      s[hlf] = __builtin_amdgcn_wmma_f32_16x16x32_f16(false, aq[1], false, b1,
                                                      (short)0, s[hlf], false, false);
    }
    const bool domask = (j0 + 31 > q0);
    #pragma unroll
    for (int hlf = 0; hlf < 2; hlf++)
      #pragma unroll
      for (int r = 0; r < 8; r++) {
        float vv = s[hlf][r] * sc;
        if (domask) {
          const int kg = j0 + hlf * 16 + ncol;
          const int qg = q0 + r + 8 * laneHi;
          if (kg > qg) vv = -__builtin_inff();
        }
        s[hlf][r] = vv;
      }
    #pragma unroll
    for (int r = 0; r < 8; r++) {
      const float tmax  = half_max16(fmaxf(s[0][r], s[1][r]));   // finite
      const float mnew  = fmaxf(mrow[r], tmax);
      const float alpha = __expf(mrow[r] - mnew);
      const float p0 = __expf(s[0][r] - mnew);
      const float p1 = __expf(s[1][r] - mnew);
      const float rsum = half_sum16(p0 + p1);
      lrow[r] = lrow[r] * alpha + rsum;
      mrow[r] = mnew;
      #pragma unroll
      for (int n = 0; n < 4; n++) oacc[n][r] *= alpha;
      const int prow = r + 8 * laneHi;           // D-layout -> LDS row-major
      pw[prow * 32 + ncol]      = (_Float16)p0;
      pw[prow * 32 + 16 + ncol] = (_Float16)p1;
    }
    v16h apf = load_a_frag(pw, 32);              // reshape P to A-fragment
    #pragma unroll
    for (int n = 0; n < 4; n++) {
      v16h bv = load_b_frag(v16 + (tokBase + j0) * E_ + headOff + n * 16, E_);
      oacc[n] = __builtin_amdgcn_wmma_f32_16x16x32_f16(false, apf, false, bv,
                                                       (short)0, oacc[n], false, false);
    }
  }
  #pragma unroll
  for (int n = 0; n < 4; n++)
    #pragma unroll
    for (int r = 0; r < 8; r++) {
      const int tok = q0 + r + 8 * laneHi;
      const float ov = oacc[n][r] / lrow[r];
      o16[(tokBase + tok) * E_ + headOff + n * 16 + ncol] = (_Float16)ov;
    }
}

// ---------------------------------------------------------------------------
extern "C" void kernel_launch(void* const* d_in, const int* in_sizes, int n_in,
                              void* d_out, int out_size, void* d_ws, size_t ws_size,
                              hipStream_t stream) {
  const int*   idx  = (const int*)  d_in[0];
  const float* tok  = (const float*)d_in[1];
  const float* pos  = (const float*)d_in[2];
  const float* ln1g = (const float*)d_in[3];
  const float* ln1b = (const float*)d_in[4];
  const float* Wq   = (const float*)d_in[5];
  const float* Wk   = (const float*)d_in[6];
  const float* Wv   = (const float*)d_in[7];
  const float* Wo   = (const float*)d_in[8];
  const float* bo   = (const float*)d_in[9];
  const float* ln2g = (const float*)d_in[10];
  const float* ln2b = (const float*)d_in[11];
  const float* W1   = (const float*)d_in[12];
  const float* b1   = (const float*)d_in[13];
  const float* W2   = (const float*)d_in[14];
  const float* b2   = (const float*)d_in[15];
  const float* lnfg = (const float*)d_in[16];
  const float* lnfb = (const float*)d_in[17];
  const float* Wlm  = (const float*)d_in[18];
  const float* blm  = (const float*)d_in[19];

  char* ws = (char*)d_ws;                                  // layout (MB offsets):
  float*    x   = (float*)    ws;                          //  0  : x   fp32  16MB
  _Float16* h16 = (_Float16*)(ws + (((size_t)16) << 20));  // 16  : h16  f16   8MB
  _Float16* q16 = (_Float16*)(ws + (((size_t)24) << 20));  // 24  : q16        8MB
  _Float16* k16 = (_Float16*)(ws + (((size_t)32) << 20));  // 32  : k16        8MB
  _Float16* v16 = (_Float16*)(ws + (((size_t)40) << 20));  // 40  : v16        8MB
  _Float16* o16 = (_Float16*)(ws + (((size_t)48) << 20));  // 48  : o16        8MB
  _Float16* u16 = (_Float16*)(ws + (((size_t)56) << 20));  // 56  : u16 (4E)  32MB

  dim3 blk(256);
  embed_k<<<M_, blk, 0, stream>>>(idx, tok, pos, x);

  const dim3 gE(E_ / 128, M_ / 128);       // N=1024
  const dim3 g1(4 * E_ / 128, M_ / 128);   // N=4096
  const dim3 gL(V_ / 128, M_ / 128);       // N=32000 (250 tiles)

  for (int l = 0; l < L_; l++) {
    layernorm_f16_k<<<M_, blk, 0, stream>>>(x, ln1g + l * E_, ln1b + l * E_, h16);
    const size_t wofs = (size_t)l * E_ * E_;
    // q/k/v projections: f16 out only
    gemm_t<false,false,false,false,true><<<gE, blk, 0, stream>>>(
        h16, Wq + wofs, nullptr, nullptr, nullptr, q16, M_, E_, E_);
    gemm_t<false,false,false,false,true><<<gE, blk, 0, stream>>>(
        h16, Wk + wofs, nullptr, nullptr, nullptr, k16, M_, E_, E_);
    gemm_t<false,false,false,false,true><<<gE, blk, 0, stream>>>(
        h16, Wv + wofs, nullptr, nullptr, nullptr, v16, M_, E_, E_);
    attn_flash_k<<<(B_ * H_ * (T_ / 16)) / 8, blk, 0, stream>>>(q16, k16, v16, o16);
    // x = x + o @ Wo + bo  (bias + residual, f32 out, in place)
    gemm_t<true,true,false,true,false><<<gE, blk, 0, stream>>>(
        o16, Wo + wofs, bo + l * E_, x, x, nullptr, M_, E_, E_);
    layernorm_f16_k<<<M_, blk, 0, stream>>>(x, ln2g + l * E_, ln2b + l * E_, h16);
    // u = relu(h2 @ W1 + b1), f16 out
    gemm_t<true,false,true,false,true><<<g1, blk, 0, stream>>>(
        h16, W1 + (size_t)l * E_ * 4 * E_, b1 + (size_t)l * 4 * E_,
        nullptr, nullptr, u16, M_, 4 * E_, E_);
    // x = x + u @ W2 + b2  (bias + residual, f32 out, in place)
    gemm_t<true,true,false,true,false><<<gE, blk, 0, stream>>>(
        u16, W2 + (size_t)l * 4 * E_ * E_, b2 + l * E_, x, x, nullptr,
        M_, E_, 4 * E_);
  }
  layernorm_f16_k<<<M_, blk, 0, stream>>>(x, lnfg, lnfb, h16);
  // logits = hf @ Wlm + blm  (bias, f32 out to d_out)
  gemm_t<true,false,false,true,false><<<gL, blk, 0, stream>>>(
      h16, Wlm, blm, nullptr, (float*)d_out, nullptr, M_, V_, E_);
}
